// MultiHeadAttentionUnit_34763465294509
// MI455X (gfx1250) — matrix operand order repaired and checked by reading, not compiled, not run on hardware
//
#include <hip/hip_runtime.h>
#include <math.h>

// ---------------------------------------------------------------------------
// MI455X (gfx1250) fused multi-head attention block.
// All GEMMs via v_wmma_f32_16x16x32_bf16 (wave32). Flash-style softmax so the
// 268MB score tensor is never materialized (L2 is 192MB; HBM would double
// traffic). Compute-bound at ~52 GFLOP vs ~85MB mandatory HBM traffic.
// ---------------------------------------------------------------------------

typedef __bf16 bf16_t;
typedef __attribute__((ext_vector_type(16))) __bf16 v16bf;
typedef __attribute__((ext_vector_type(8)))  __bf16 v8bf;
typedef __attribute__((ext_vector_type(8)))  float  v8f;

#define B_  4
#define S_  1024
#define M_  1024
#define H_  16
#define K_  64
#define BS_ 4096   // B*S
#define HV_ 1024   // H*V

__device__ __forceinline__ v8f wmma_bf16(v16bf a, v16bf b, v8f c) {
  // D = A(16x32) * B(32x16) + C(16x16 f32)
  return __builtin_amdgcn_wmma_f32_16x16x32_bf16(false, a, false, b, (short)0, c,
                                                 false, false);
}

// A-operand fragment (16x32 bf16). Lane L holds row (L&15); K permutation per
// ISA table: lanes<16: e0..7 -> K0..7, e8..15 -> K16..23; lanes>=16: +8.
__device__ __forceinline__ v16bf load_frag_A(const bf16_t* row, int hi) {
  v8bf p0 = *(const v8bf*)(row + 8 * hi);
  v8bf p1 = *(const v8bf*)(row + 16 + 8 * hi);
  v16bf f;
#pragma unroll
  for (int i = 0; i < 8; ++i) { f[i] = p0[i]; f[i + 8] = p1[i]; }
  return f;
}

// B-operand fragment (32x16 bf16), source stored row-major per output column n
// (contraction contiguous): same K permutation as A (e0..7 -> K0..7+8*hi, e8..15 -> K16..23+8*hi).
__device__ __forceinline__ v16bf load_frag_B(const bf16_t* row, int hi) {
  v8bf p0 = *(const v8bf*)(row + 8 * hi);
  v8bf p1 = *(const v8bf*)(row + 16 + 8 * hi);
  v16bf f;
#pragma unroll
  for (int i = 0; i < 8; ++i) { f[i] = p0[i]; f[i + 8] = p1[i]; }
  return f;
}

// reductions within the 16-lane halves (matches WMMA C layout: col = lane&15)
__device__ __forceinline__ float halfmax(float v) {
#pragma unroll
  for (int off = 1; off < 16; off <<= 1) v = fmaxf(v, __shfl_xor(v, off, 32));
  return v;
}
__device__ __forceinline__ float halfsum(float v) {
#pragma unroll
  for (int off = 1; off < 16; off <<= 1) v += __shfl_xor(v, off, 32);
  return v;
}

// --------------------------- prep kernels ----------------------------------
__global__ __launch_bounds__(256) void cast_bf16_kernel(
    const float* __restrict__ src, bf16_t* __restrict__ dst, int n4) {
  int i = blockIdx.x * 256 + threadIdx.x;
  if (i < n4) {
    float4 f = ((const float4*)src)[i];
    int idx = i * 4;
    dst[idx + 0] = (bf16_t)f.x; dst[idx + 1] = (bf16_t)f.y;
    dst[idx + 2] = (bf16_t)f.z; dst[idx + 3] = (bf16_t)f.w;
  }
}

// w[h][m][k] (f32) -> wT[h][k][m] (bf16); i runs over H*K*M, z selects array.
__global__ __launch_bounds__(256) void wtrans_kernel(
    const float* __restrict__ wq, const float* __restrict__ wk,
    const float* __restrict__ wv, bf16_t* __restrict__ oq,
    bf16_t* __restrict__ ok2, bf16_t* __restrict__ ov) {
  const float* in = (blockIdx.z == 0) ? wq : (blockIdx.z == 1) ? wk : wv;
  bf16_t* out = (blockIdx.z == 0) ? oq : (blockIdx.z == 1) ? ok2 : ov;
  int i = blockIdx.x * 256 + threadIdx.x;        // over H*K*M
  int m = i & (M_ - 1);
  int k = (i >> 10) & (K_ - 1);
  int h = i >> 16;
  out[i] = (bf16_t)in[((size_t)h * M_ + m) * K_ + k];
}

// --------------------------- projections -----------------------------------
// C[4096,64] = X[4096,1024] * Wt[h][64,1024]^T  per head, per z in {q,k,v}.
// z<2: store [h][row][k]; z==2: store transposed mvT[h][b][v][s].
__global__ __launch_bounds__(256) void proj_kernel(
    const bf16_t* __restrict__ qb, const bf16_t* __restrict__ kb,
    const bf16_t* __restrict__ vb, const bf16_t* __restrict__ wqT,
    const bf16_t* __restrict__ wkT, const bf16_t* __restrict__ wvT,
    bf16_t* __restrict__ mq, bf16_t* __restrict__ mk,
    bf16_t* __restrict__ mvT) {
  const int z = blockIdx.z, h = blockIdx.y;
  const bf16_t* X  = (z == 0) ? qb  : (z == 1) ? kb  : vb;
  const bf16_t* Wt = (z == 0) ? wqT : (z == 1) ? wkT : wvT;
  const int lane = threadIdx.x & 31, wave = threadIdx.x >> 5;
  const int hi = lane >> 4, ln = lane & 15;
  const int rbase = blockIdx.x * 128 + wave * 16;
  const bf16_t* arow = X + (size_t)(rbase + ln) * M_;
  const bf16_t* wb = Wt + (size_t)h * K_ * M_;

  v8f acc[4];
#pragma unroll
  for (int t = 0; t < 4; ++t)
#pragma unroll
    for (int r = 0; r < 8; ++r) acc[t][r] = 0.f;

  for (int kc = 0; kc < M_; kc += 32) {
    v16bf a = load_frag_A(arow + kc, hi);
#pragma unroll
    for (int t = 0; t < 4; ++t) {
      v16bf bfrag = load_frag_B(wb + (size_t)(t * 16 + ln) * M_ + kc, hi);
      acc[t] = wmma_bf16(a, bfrag, acc[t]);
    }
  }

  if (z < 2) {
    bf16_t* out = ((z == 0) ? mq : mk) + (size_t)h * BS_ * K_;
#pragma unroll
    for (int t = 0; t < 4; ++t)
#pragma unroll
      for (int r = 0; r < 8; ++r) {
        int grow = rbase + r + 8 * hi;
        out[(size_t)grow * K_ + t * 16 + ln] = (bf16_t)acc[t][r];
      }
  } else {
#pragma unroll
    for (int t = 0; t < 4; ++t)
#pragma unroll
      for (int r = 0; r < 8; ++r) {
        int grow = rbase + r + 8 * hi;
        int bb = grow >> 10, ss = grow & 1023;
        mvT[(((size_t)h * B_ + bb) * K_ + (t * 16 + ln)) * S_ + ss] =
            (bf16_t)acc[t][r];
      }
  }
}

// --------------------------- fused flash attention -------------------------
// grid (S/64, B, H), block 128 (4 waves); wave handles 16 q rows x full V.
__global__ __launch_bounds__(128) void attn_kernel(
    const bf16_t* __restrict__ mq, const bf16_t* __restrict__ mk,
    const bf16_t* __restrict__ mvT, const unsigned char* __restrict__ mask,
    bf16_t* __restrict__ ctx) {
  const int h = blockIdx.z, b = blockIdx.y;
  const int lane = threadIdx.x & 31, wave = threadIdx.x >> 5;
  const int hi = lane >> 4, ln = lane & 15;
  const int qbase = blockIdx.x * 64 + wave * 16;

  const size_t hb = (size_t)h * B_ + b;
  const bf16_t* mqh = mq + hb * (size_t)S_ * K_;
  const bf16_t* mkh = mk + hb * (size_t)S_ * K_;
  const bf16_t* mvh = mvT + hb * (size_t)K_ * S_;
  const unsigned char* mb = mask + (size_t)b * S_ * S_;

  // Persistent A fragments for this wave's 16 q rows (K = 64 -> 2 chunks)
  const v16bf aq0 = load_frag_A(mqh + (size_t)(qbase + ln) * K_ + 0, hi);
  const v16bf aq1 = load_frag_A(mqh + (size_t)(qbase + ln) * K_ + 32, hi);

  float mrow[8], lrow[8];
  v8f o[4];
#pragma unroll
  for (int r = 0; r < 8; ++r) { mrow[r] = -1e30f; lrow[r] = 0.f; }
#pragma unroll
  for (int t = 0; t < 4; ++t)
#pragma unroll
    for (int r = 0; r < 8; ++r) o[t][r] = 0.f;

  __shared__ bf16_t pbuf[4][16 * 32];     // per-wave P tile (C->A relayout)
  bf16_t* myp = pbuf[wave];
  const float invScale = 1.0f / 32.0f;    // 1/sqrt(MODEL_DIM)

  for (int s0 = 0; s0 < S_; s0 += 32) {
    v8f sc0, sc1;
#pragma unroll
    for (int r = 0; r < 8; ++r) { sc0[r] = 0.f; sc1[r] = 0.f; }
    {
      v16bf b0 = load_frag_B(mkh + (size_t)(s0 + ln) * K_ + 0, hi);
      v16bf b1 = load_frag_B(mkh + (size_t)(s0 + ln) * K_ + 32, hi);
      sc0 = wmma_bf16(aq0, b0, sc0);
      sc0 = wmma_bf16(aq1, b1, sc0);
      v16bf b2 = load_frag_B(mkh + (size_t)(s0 + 16 + ln) * K_ + 0, hi);
      v16bf b3 = load_frag_B(mkh + (size_t)(s0 + 16 + ln) * K_ + 32, hi);
      sc1 = wmma_bf16(aq0, b2, sc1);
      sc1 = wmma_bf16(aq1, b3, sc1);
    }
    float p0[8], p1[8];
#pragma unroll
    for (int r = 0; r < 8; ++r) {
      const int qr = qbase + r + 8 * hi;
      float v0 = sc0[r] * invScale;
      float v1 = sc1[r] * invScale;
      if (mb[(size_t)qr * S_ + s0 + ln])      v0 = -1e30f;
      if (mb[(size_t)qr * S_ + s0 + 16 + ln]) v1 = -1e30f;
      float rm = halfmax(fmaxf(v0, v1));
      float mn = fmaxf(mrow[r], rm);
      float corr = __expf(mrow[r] - mn);
      float e0 = __expf(v0 - mn);
      float e1 = __expf(v1 - mn);
      lrow[r] = lrow[r] * corr + halfsum(e0 + e1);
      mrow[r] = mn;
      p0[r] = e0; p1[r] = e1;
#pragma unroll
      for (int t = 0; t < 4; ++t) o[t][r] *= corr;
    }
    // C-layout -> A-layout via per-wave LDS tile (wave-local: dscnt only)
#pragma unroll
    for (int r = 0; r < 8; ++r) {
      myp[(r + 8 * hi) * 32 + ln]      = (bf16_t)p0[r];
      myp[(r + 8 * hi) * 32 + 16 + ln] = (bf16_t)p1[r];
    }
    asm volatile("s_wait_dscnt 0" ::: "memory");
    v16bf ap = load_frag_A(myp + ln * 32, hi);
#pragma unroll
    for (int t = 0; t < 4; ++t) {
      v16bf bv = load_frag_B(mvh + (size_t)(t * 16 + ln) * S_ + s0, hi);
      o[t] = wmma_bf16(ap, bv, o[t]);
    }
    asm volatile("s_wait_dscnt 0" ::: "memory"); // ap consumed before rewrite
  }

#pragma unroll
  for (int t = 0; t < 4; ++t)
#pragma unroll
    for (int r = 0; r < 8; ++r) {
      const int qr = qbase + r + 8 * hi;
      float val = o[t][r] / lrow[r];
      ctx[((size_t)b * S_ + qr) * HV_ + h * K_ + t * 16 + ln] = (bf16_t)val;
    }
}

// --------------------------- output linear + residual ----------------------
__global__ __launch_bounds__(256) void linear_kernel(
    const bf16_t* __restrict__ ctx, const bf16_t* __restrict__ linWb,
    const float* __restrict__ lin_b, const float* __restrict__ resid,
    float* __restrict__ linout) {
  const int lane = threadIdx.x & 31, wave = threadIdx.x >> 5;
  const int hi = lane >> 4, ln = lane & 15;
  const int rbase = blockIdx.x * 128 + wave * 16;
  const int cbase = blockIdx.y * 64;
  const bf16_t* arow = ctx + (size_t)(rbase + ln) * HV_;

  v8f acc[4];
#pragma unroll
  for (int t = 0; t < 4; ++t)
#pragma unroll
    for (int r = 0; r < 8; ++r) acc[t][r] = 0.f;

  for (int kc = 0; kc < HV_; kc += 32) {
    v16bf a = load_frag_A(arow + kc, hi);
#pragma unroll
    for (int t = 0; t < 4; ++t) {
      v16bf bfrag =
          load_frag_B(linWb + (size_t)(cbase + t * 16 + ln) * HV_ + kc, hi);
      acc[t] = wmma_bf16(a, bfrag, acc[t]);
    }
  }
#pragma unroll
  for (int t = 0; t < 4; ++t)
#pragma unroll
    for (int r = 0; r < 8; ++r) {
      int grow = rbase + r + 8 * hi;
      int col = cbase + t * 16 + ln;
      linout[(size_t)grow * M_ + col] =
          acc[t][r] + lin_b[col] + resid[(size_t)grow * M_ + col];
    }
}

// --------------------------- layernorm (Bessel var, eps on sigma) ----------
__global__ __launch_bounds__(256) void ln_kernel(
    const float* __restrict__ x, const float* __restrict__ a2,
    const float* __restrict__ b2, float* __restrict__ out) {
  const int row = blockIdx.x;
  const float* xr = x + (size_t)row * M_;
  float v[4], s = 0.f, s2 = 0.f;
#pragma unroll
  for (int i = 0; i < 4; ++i) {
    v[i] = xr[threadIdx.x + i * 256];
    s += v[i]; s2 += v[i] * v[i];
  }
  __shared__ float red0[256], red1[256];
  red0[threadIdx.x] = s; red1[threadIdx.x] = s2;
  __syncthreads();
  for (int off = 128; off > 0; off >>= 1) {
    if ((int)threadIdx.x < off) {
      red0[threadIdx.x] += red0[threadIdx.x + off];
      red1[threadIdx.x] += red1[threadIdx.x + off];
    }
    __syncthreads();
  }
  const float mu = red0[0] * (1.0f / M_);
  const float var = (red1[0] - (float)M_ * mu * mu) * (1.0f / (M_ - 1));
  const float inv = 1.0f / (sqrtf(fmaxf(var, 0.f)) + 1e-3f);
  float* orow = out + (size_t)row * M_;
#pragma unroll
  for (int i = 0; i < 4; ++i) {
    int j = threadIdx.x + i * 256;
    orow[j] = (v[i] - mu) * inv * a2[j] + b2[j];
  }
}

// --------------------------- host ------------------------------------------
extern "C" void kernel_launch(void* const* d_in, const int* in_sizes, int n_in,
                              void* d_out, int out_size, void* d_ws,
                              size_t ws_size, hipStream_t stream) {
  const float* q = (const float*)d_in[0];
  const float* k = (const float*)d_in[1];
  const float* v = (const float*)d_in[2];
  const unsigned char* mask = (const unsigned char*)d_in[3];
  const float* w_qs = (const float*)d_in[4];
  const float* w_ks = (const float*)d_in[5];
  const float* w_vs = (const float*)d_in[6];
  const float* lin_w = (const float*)d_in[7];
  const float* lin_b = (const float*)d_in[8];
  const float* a2 = (const float*)d_in[9];
  const float* b2 = (const float*)d_in[10];
  float* out = (float*)d_out;

  char* ws = (char*)d_ws;
  size_t off = 0;
  auto alloc = [&](size_t bytes) -> void* {
    void* p = ws + off;
    off = (off + bytes + 255) & ~(size_t)255;
    return p;
  };
  bf16_t* qb    = (bf16_t*)alloc((size_t)BS_ * M_ * 2);
  bf16_t* kb    = (bf16_t*)alloc((size_t)BS_ * M_ * 2);
  bf16_t* vb    = (bf16_t*)alloc((size_t)BS_ * M_ * 2);
  bf16_t* wqT   = (bf16_t*)alloc((size_t)H_ * K_ * M_ * 2);
  bf16_t* wkT   = (bf16_t*)alloc((size_t)H_ * K_ * M_ * 2);
  bf16_t* wvT   = (bf16_t*)alloc((size_t)H_ * K_ * M_ * 2);
  bf16_t* linWb = (bf16_t*)alloc((size_t)M_ * HV_ * 2);
  bf16_t* mq    = (bf16_t*)alloc((size_t)H_ * BS_ * K_ * 2);
  bf16_t* mk    = (bf16_t*)alloc((size_t)H_ * BS_ * K_ * 2);
  bf16_t* mvT   = (bf16_t*)alloc((size_t)H_ * BS_ * K_ * 2);
  bf16_t* ctx   = (bf16_t*)alloc((size_t)BS_ * HV_ * 2);
  float*  lino  = (float*) alloc((size_t)BS_ * M_ * 4);

  // 1. bf16 casts
  cast_bf16_kernel<<<(BS_ * M_ / 4 + 255) / 256, 256, 0, stream>>>(q, qb, BS_ * M_ / 4);
  cast_bf16_kernel<<<(BS_ * M_ / 4 + 255) / 256, 256, 0, stream>>>(k, kb, BS_ * M_ / 4);
  cast_bf16_kernel<<<(BS_ * M_ / 4 + 255) / 256, 256, 0, stream>>>(v, vb, BS_ * M_ / 4);
  cast_bf16_kernel<<<(M_ * HV_ / 4 + 255) / 256, 256, 0, stream>>>(lin_w, linWb, M_ * HV_ / 4);

  // 2. per-head weight transpose-cast: [H,M,K] -> [H,K,M] bf16
  wtrans_kernel<<<dim3(H_ * K_ * M_ / 256, 1, 3), 256, 0, stream>>>(
      w_qs, w_ks, w_vs, wqT, wkT, wvT);

  // 3. q/k/v projections (WMMA)
  proj_kernel<<<dim3(BS_ / 128, H_, 3), 256, 0, stream>>>(
      qb, kb, vb, wqT, wkT, wvT, mq, mk, mvT);

  // 4. fused flash attention (WMMA scores + online softmax + WMMA P*V)
  attn_kernel<<<dim3(S_ / 64, B_, H_), 128, 0, stream>>>(mq, mk, mvT, mask, ctx);

  // 5. output linear + bias + residual (WMMA)
  linear_kernel<<<dim3(BS_ / 128, M_ / 64), 256, 0, stream>>>(
      ctx, linWb, lin_b, q, lino);

  // 6. layernorm
  ln_kernel<<<BS_, 256, 0, stream>>>(lino, a2, b2, out);
}
